// GA_Module_4672924418089
// MI455X (gfx1250) — matrix operand order repaired and checked by reading, not compiled, not run on hardware
//
#include <hip/hip_runtime.h>
#include <math.h>
#include <stdint.h>

// Problem sizes (fixed by the reference)
#define B_  8
#define C_  512
#define N_  2048
#define D_  512   // QK == V == 512

typedef __attribute__((ext_vector_type(8)))  _Float16 v8h;
typedef __attribute__((ext_vector_type(16))) _Float16 v16h;
typedef __attribute__((ext_vector_type(8)))  float    v8f;
typedef __attribute__((ext_vector_type(4)))  int      v4i;

typedef __attribute__((address_space(1))) v4i* gv4i_ptr;   // global int4*
typedef __attribute__((address_space(3))) v4i* lv4i_ptr;   // LDS int4*

// ---------------------------------------------------------------------------
// CDNA5 WMMA fragment loaders (V_WMMA_F32_16X16X32_F16 layouts, wave32)
//
// A (16x32, 16-bit): lanes 0-15 hold M=0..15 / K = kk+0..7 (halves 0-7) and
// K = kk+16..23 (halves 8-15); lanes 16-31 hold K = kk+8..15 / kk+24..31.
// Memory layout assumed: element (row, k) at base[row*ld + k] (K contiguous).
__device__ __forceinline__ v16h ld_a16(const _Float16* base, int row0, int ld,
                                       int kk, int lane) {
  const int hs  = lane >> 4;
  const int row = row0 + (lane & 15);
  const _Float16* p = base + (size_t)row * ld + kk + hs * 8;
  v8h lo = *(const v8h*)p;          // 16B load
  v8h hi = *(const v8h*)(p + 16);   // 16B load
  v16h a;
#pragma unroll
  for (int i = 0; i < 8; ++i) { a[i] = lo[i]; a[i + 8] = hi[i]; }
  return a;
}

// B (32x16, 16-bit): lane L<16 -> column col0+L, K = kk+0..7 / kk+16..23;
// lanes 16-31 -> K = kk+8..15 / kk+24..31. Memory: element (k, col) at base[col*ld + k].
__device__ __forceinline__ v16h ld_b16(const _Float16* base, int col0, int ld,
                                       int kk, int lane) {
  const int col = col0 + (lane & 15);
  const int ko  = kk + (lane >> 4) * 8;
  const _Float16* p = base + (size_t)col * ld + ko;
  v8h lo = *(const v8h*)p;
  v8h hi = *(const v8h*)(p + 16);
  v16h r;
#pragma unroll
  for (int i = 0; i < 8; ++i) { r[i] = lo[i]; r[i + 8] = hi[i]; }
  return r;
}

// ---------------------------------------------------------------------------
// Kernel 1: convert weights f32 -> f16; fold 1/sqrt(D) into W_q.
__global__ void wconv_kernel(const float* __restrict__ wq,
                             const float* __restrict__ wk,
                             const float* __restrict__ wv,
                             _Float16* __restrict__ wqh,
                             _Float16* __restrict__ wkh,
                             _Float16* __restrict__ wvh, float invT) {
  int i = blockIdx.x * blockDim.x + threadIdx.x;
  if (i < D_ * C_) {
    wqh[i] = (_Float16)(wq[i] * invT);
    wkh[i] = (_Float16)wk[i];
    wvh[i] = (_Float16)wv[i];
  }
}

// ---------------------------------------------------------------------------
// Kernel 2: x [B,C,N] f32 -> Xt [B,N,C] f16 (LDS tile transpose + convert)
__global__ void xt_kernel(const float* __restrict__ x, _Float16* __restrict__ xt) {
  __shared__ float tile[32][33];
  const int b  = blockIdx.z;
  const int n0 = blockIdx.x * 32;
  const int c0 = blockIdx.y * 32;
  const float* xb = x + (size_t)b * C_ * N_;
  _Float16* xtb = xt + (size_t)b * N_ * C_;
#pragma unroll
  for (int k = 0; k < 4; ++k) {
    int c = c0 + threadIdx.y + k * 8;
    tile[threadIdx.y + k * 8][threadIdx.x] = xb[(size_t)c * N_ + n0 + threadIdx.x];
  }
  __syncthreads();
#pragma unroll
  for (int k = 0; k < 4; ++k) {
    int n = n0 + threadIdx.y + k * 8;
    xtb[(size_t)n * C_ + c0 + threadIdx.x] = (_Float16)tile[threadIdx.x][threadIdx.y + k * 8];
  }
}

// ---------------------------------------------------------------------------
// Kernel 3: Q/K projection.  out[b,n,o] = sum_c Xt[b,n,c] * W[o,c]   (f16 out)
// Each wave: one 16(n) x 64(o) tile, K = 512. Software-pipelined A loads.
__global__ __launch_bounds__(256) void projqk_kernel(const _Float16* __restrict__ xt,
                                                     const _Float16* __restrict__ wh,
                                                     _Float16* __restrict__ outp) {
  const int lane = threadIdx.x & 31;
  const int w    = threadIdx.x >> 5;
  const int b    = blockIdx.z;
  const int n0   = (blockIdx.x * 8 + w) * 16;
  const int o0   = blockIdx.y * 64;
  const _Float16* xtb = xt + (size_t)b * N_ * C_;
  _Float16* ob = outp + (size_t)b * N_ * D_;
  v8f acc[4] = {};
  v16h a = ld_a16(xtb, n0, C_, 0, lane);
#pragma unroll
  for (int kc = 0; kc < C_ / 32; ++kc) {
    v16h an = a;
    if (kc + 1 < C_ / 32) an = ld_a16(xtb, n0, C_, (kc + 1) * 32, lane);
    v16h b0 = ld_b16(wh, o0 +  0, C_, kc * 32, lane);
    v16h b1 = ld_b16(wh, o0 + 16, C_, kc * 32, lane);
    v16h b2 = ld_b16(wh, o0 + 32, C_, kc * 32, lane);
    v16h b3 = ld_b16(wh, o0 + 48, C_, kc * 32, lane);
    acc[0] = __builtin_amdgcn_wmma_f32_16x16x32_f16(false, a, false, b0, (short)0, acc[0], false, false);
    acc[1] = __builtin_amdgcn_wmma_f32_16x16x32_f16(false, a, false, b1, (short)0, acc[1], false, false);
    acc[2] = __builtin_amdgcn_wmma_f32_16x16x32_f16(false, a, false, b2, (short)0, acc[2], false, false);
    acc[3] = __builtin_amdgcn_wmma_f32_16x16x32_f16(false, a, false, b3, (short)0, acc[3], false, false);
    a = an;
  }
  // D layout: lane -> col (lane&15), VGPR r -> row r + 8*(lane>>4)
#pragma unroll
  for (int ot = 0; ot < 4; ++ot)
#pragma unroll
    for (int r = 0; r < 8; ++r) {
      int n = n0 + r + 8 * (lane >> 4);
      int o = o0 + ot * 16 + (lane & 15);
      ob[(size_t)n * D_ + o] = (_Float16)acc[ot][r];
    }
}

// ---------------------------------------------------------------------------
// Kernel 4: V projection.  Vh[b,v,n] = sum_c Wv[v,c] * Xt[b,n,c] + bv[v]
// Each wave: one 16(v) x 64(n) tile.
__global__ __launch_bounds__(256) void projv_kernel(const _Float16* __restrict__ xt,
                                                    const _Float16* __restrict__ wvh,
                                                    const float* __restrict__ bv,
                                                    _Float16* __restrict__ vh) {
  const int lane = threadIdx.x & 31;
  const int w    = threadIdx.x >> 5;
  const int b    = blockIdx.z;
  const int v0   = (blockIdx.y * 8 + w) * 16;
  const int n0   = blockIdx.x * 64;
  const _Float16* xtb = xt + (size_t)b * N_ * C_;
  _Float16* vb = vh + (size_t)b * D_ * N_;
  v8f acc[4] = {};
  v16h a = ld_a16(wvh, v0, C_, 0, lane);
#pragma unroll
  for (int kc = 0; kc < C_ / 32; ++kc) {
    v16h an = a;
    if (kc + 1 < C_ / 32) an = ld_a16(wvh, v0, C_, (kc + 1) * 32, lane);
    v16h b0 = ld_b16(xtb, n0 +  0, C_, kc * 32, lane);
    v16h b1 = ld_b16(xtb, n0 + 16, C_, kc * 32, lane);
    v16h b2 = ld_b16(xtb, n0 + 32, C_, kc * 32, lane);
    v16h b3 = ld_b16(xtb, n0 + 48, C_, kc * 32, lane);
    acc[0] = __builtin_amdgcn_wmma_f32_16x16x32_f16(false, a, false, b0, (short)0, acc[0], false, false);
    acc[1] = __builtin_amdgcn_wmma_f32_16x16x32_f16(false, a, false, b1, (short)0, acc[1], false, false);
    acc[2] = __builtin_amdgcn_wmma_f32_16x16x32_f16(false, a, false, b2, (short)0, acc[2], false, false);
    acc[3] = __builtin_amdgcn_wmma_f32_16x16x32_f16(false, a, false, b3, (short)0, acc[3], false, false);
    a = an;
  }
  float bias[8];
#pragma unroll
  for (int r = 0; r < 8; ++r) bias[r] = bv[v0 + r + 8 * (lane >> 4)];
#pragma unroll
  for (int nt = 0; nt < 4; ++nt)
#pragma unroll
    for (int r = 0; r < 8; ++r) {
      int v = v0 + r + 8 * (lane >> 4);
      int n = n0 + nt * 16 + (lane & 15);
      vb[(size_t)v * N_ + n] = (_Float16)(acc[nt][r] + bias[r]);
    }
}

// ---------------------------------------------------------------------------
// Kernel 5: fused flash attention.
// One workgroup = (batch b, 16 query rows). 8 waves / 256 threads.
// Iterate keys in chunks of 128; wave w computes S m-tile w, and owns
// output channels v in [w*64, w*64+64).
__global__ __launch_bounds__(256) void attn_kernel(const _Float16* __restrict__ Qt,
                                                   const _Float16* __restrict__ Kt,
                                                   const _Float16* __restrict__ Vh,
                                                   float* __restrict__ out) {
  __shared__ float    sS[16][128];          // 8 KB: S chunk (already /T via Q scale)
  __shared__ _Float16 sP[16][128];          // 4 KB: P = exp(S - max) in f16
  __shared__ float    red[16][16];          // 1 KB: per-row partial reductions
  __shared__ float    rowmax[16], rowsum[16], rowscale[16];
  __shared__ float    sO[8][64][16];        // 32 KB: output staging (per wave 64v x 16n)

  const int t    = threadIdx.x;
  const int lane = t & 31;
  const int w    = t >> 5;
  const int b    = blockIdx.y;
  const int n0   = blockIdx.x * 16;

  const _Float16* Qb = Qt + (size_t)b * N_ * D_;
  const _Float16* Kb = Kt + (size_t)b * N_ * D_;
  const _Float16* Vb = Vh + (size_t)b * D_ * N_;

  // Q A-fragments for the whole 512-deep K dimension, kept in registers.
  v16h aq[16];
#pragma unroll
  for (int kc = 0; kc < 16; ++kc) aq[kc] = ld_a16(Qb, n0, D_, kc * 32, lane);

  v8f o[4] = {};   // O accumulators: 4 v-tiles of 16 cols each (64 channels/wave)

  if (t < 16) { rowmax[t] = -3.0e38f; rowsum[t] = 0.0f; }
  __syncthreads();

  const int row = t & 15;   // softmax: thread -> (row, 8-col segment)
  const int seg = t >> 4;
  const int v0  = w * 64;

  for (int it = 0; it < 16; ++it) {
    const int mbase = it * 128;

    // ---- prefetch next chunk's K rows / V columns into near caches
    if (it + 1 < 16) {
      const int mn = mbase + 128;
      const _Float16* kp = Kb + (size_t)(mn + w * 16 + (lane & 15)) * D_ + (lane >> 4) * 64;
#pragma unroll
      for (int pf = 0; pf < 4; ++pf) __builtin_prefetch(kp + pf * 128, 0, 3);
#pragma unroll
      for (int pr = 0; pr < 2; ++pr)
#pragma unroll
        for (int pc = 0; pc < 2; ++pc)
          __builtin_prefetch(Vb + (size_t)(v0 + lane + pr * 32) * N_ + mn + pc * 64, 0, 3);
    }

    // ---- S tile: wave w computes rows n0..n0+15 x cols mbase+w*16..+15
    {
      v8f s = {};
      const int mt = mbase + w * 16;
      v16h bk = ld_b16(Kb, mt, D_, 0, lane);
#pragma unroll
      for (int kc = 0; kc < 16; ++kc) {
        v16h bn = bk;
        if (kc + 1 < 16) bn = ld_b16(Kb, mt, D_, (kc + 1) * 32, lane);
        s = __builtin_amdgcn_wmma_f32_16x16x32_f16(false, aq[kc], false, bk,
                                                   (short)0, s, false, false);
        bk = bn;
      }
#pragma unroll
      for (int r = 0; r < 8; ++r)
        sS[r + 8 * (lane >> 4)][w * 16 + (lane & 15)] = s[r];
    }
    __syncthreads();

    // ---- online softmax over the 16x128 chunk
    float mloc = -3.0e38f;
#pragma unroll
    for (int j = 0; j < 8; ++j) mloc = fmaxf(mloc, sS[row][seg * 8 + j]);
    red[row][seg] = mloc;
    __syncthreads();
    if (t < 16) {
      float cm = red[t][0];
      for (int j = 1; j < 16; ++j) cm = fmaxf(cm, red[t][j]);
      float om = rowmax[t];
      float nm = fmaxf(om, cm);
      rowmax[t]   = nm;
      rowscale[t] = __expf(om - nm);   // == 0 on first chunk
    }
    __syncthreads();
    {
      float nm = rowmax[row];
      float ps = 0.0f;
#pragma unroll
      for (int j = 0; j < 8; ++j) {
        float e = __expf(sS[row][seg * 8 + j] - nm);
        sP[row][seg * 8 + j] = (_Float16)e;
        ps += e;
      }
      red[row][seg] = ps;
    }
    __syncthreads();
    if (t < 16) {
      float cs = 0.0f;
      for (int j = 0; j < 16; ++j) cs += red[t][j];
      rowsum[t] = rowsum[t] * rowscale[t] + cs;
    }
    __syncthreads();

    // ---- rescale O accumulators, then O += P * V^T for this wave's 64 channels
    {
      float sc[8];
#pragma unroll
      for (int r = 0; r < 8; ++r) sc[r] = rowscale[r + 8 * (lane >> 4)];
#pragma unroll
      for (int vt = 0; vt < 4; ++vt)
#pragma unroll
        for (int r = 0; r < 8; ++r) o[vt][r] *= sc[r];

      v16h ap = ld_a16(&sP[0][0], 0, 128, 0, lane);   // LDS -> ds_load_b128
#pragma unroll
      for (int kc = 0; kc < 4; ++kc) {
        v16h apn = ap;
        if (kc + 1 < 4) apn = ld_a16(&sP[0][0], 0, 128, (kc + 1) * 32, lane);
        v16h b0 = ld_b16(Vb, v0 +  0, N_, mbase + kc * 32, lane);
        v16h b1 = ld_b16(Vb, v0 + 16, N_, mbase + kc * 32, lane);
        v16h b2 = ld_b16(Vb, v0 + 32, N_, mbase + kc * 32, lane);
        v16h b3 = ld_b16(Vb, v0 + 48, N_, mbase + kc * 32, lane);
        o[0] = __builtin_amdgcn_wmma_f32_16x16x32_f16(false, ap, false, b0, (short)0, o[0], false, false);
        o[1] = __builtin_amdgcn_wmma_f32_16x16x32_f16(false, ap, false, b1, (short)0, o[1], false, false);
        o[2] = __builtin_amdgcn_wmma_f32_16x16x32_f16(false, ap, false, b2, (short)0, o[2], false, false);
        o[3] = __builtin_amdgcn_wmma_f32_16x16x32_f16(false, ap, false, b3, (short)0, o[3], false, false);
        ap = apn;
      }
    }
    __syncthreads();   // protect sS/sP/rowscale before next chunk
  }

  // ---- finalize: divide by row sums, stage via LDS, bulk store
  float inv[8];
#pragma unroll
  for (int r = 0; r < 8; ++r) inv[r] = 1.0f / rowsum[r + 8 * (lane >> 4)];
#pragma unroll
  for (int vt = 0; vt < 4; ++vt)
#pragma unroll
    for (int r = 0; r < 8; ++r)
      sO[w][vt * 16 + (lane & 15)][r + 8 * (lane >> 4)] = o[vt][r] * inv[r];
  __syncthreads();

#if __has_builtin(__builtin_amdgcn_global_store_async_from_lds_b128) && \
    __has_builtin(__builtin_amdgcn_s_wait_asynccnt)
  // Async LDS -> global bulk store: each lane moves 16B chunks (4 floats of
  // one v-row); 8 instructions cover the wave's 64x16 fp32 tile.
  // AS1 pointer bits == 64-bit VA; AS3 pointer bits == low-32 LDS offset,
  // so integer casts produce exactly the right pointer values.
#pragma unroll
  for (int i = 0; i < 8; ++i) {
    int idx = i * 32 + lane;        // 16B-chunk index within wave tile
    int vv  = idx >> 2;             // 4 chunks per v-row of 16 floats
    int nn  = (idx & 3) * 4;
    float* gp = out + ((size_t)b * D_ + v0 + vv) * N_ + n0 + nn;
    __builtin_amdgcn_global_store_async_from_lds_b128(
        (gv4i_ptr)(uintptr_t)gp,
        (lv4i_ptr)(uint32_t)(uintptr_t)&sO[w][vv][nn], 0, 0);
  }
  __builtin_amdgcn_s_wait_asynccnt(0);
#else
  for (int i = lane; i < 64 * 16; i += 32) {
    int vv = i >> 4, nn = i & 15;
    out[((size_t)b * D_ + v0 + vv) * N_ + n0 + nn] = sO[w][vv][nn];
  }
#endif
}

// ---------------------------------------------------------------------------
extern "C" void kernel_launch(void* const* d_in, const int* in_sizes, int n_in,
                              void* d_out, int out_size, void* d_ws, size_t ws_size,
                              hipStream_t stream) {
  const float* x  = (const float*)d_in[0];
  const float* wq = (const float*)d_in[1];
  const float* wk = (const float*)d_in[2];
  const float* wv = (const float*)d_in[3];
  const float* bv = (const float*)d_in[4];
  float* out = (float*)d_out;

  char* ws = (char*)d_ws;
  size_t off = 0;
  _Float16* Wqh = (_Float16*)(ws + off); off += (size_t)D_ * C_ * 2;
  _Float16* Wkh = (_Float16*)(ws + off); off += (size_t)D_ * C_ * 2;
  _Float16* Wvh = (_Float16*)(ws + off); off += (size_t)D_ * C_ * 2;
  _Float16* Xt  = (_Float16*)(ws + off); off += (size_t)B_ * N_ * C_ * 2;
  _Float16* Qt  = (_Float16*)(ws + off); off += (size_t)B_ * N_ * D_ * 2;
  _Float16* Kt  = (_Float16*)(ws + off); off += (size_t)B_ * N_ * D_ * 2;
  _Float16* Vh  = (_Float16*)(ws + off); off += (size_t)B_ * D_ * N_ * 2;
  (void)ws_size; (void)in_sizes; (void)n_in; (void)out_size;

  const float invT = 1.0f / sqrtf((float)D_);   // fold 1/sqrt(512) into W_q

  wconv_kernel<<<(D_ * C_ + 255) / 256, 256, 0, stream>>>(wq, wk, wv, Wqh, Wkh, Wvh, invT);
  xt_kernel<<<dim3(N_ / 32, C_ / 32, B_), dim3(32, 8), 0, stream>>>(x, Xt);
  projqk_kernel<<<dim3(N_ / 128, D_ / 64, B_), 256, 0, stream>>>(Xt, Wqh, Qt);
  projqk_kernel<<<dim3(N_ / 128, D_ / 64, B_), 256, 0, stream>>>(Xt, Wkh, Kt);
  projv_kernel<<<dim3(N_ / 64, D_ / 128, B_), 256, 0, stream>>>(Xt, Wvh, bv, Vh);
  attn_kernel<<<dim3(N_ / 16, B_), 256, 0, stream>>>(Qt, Kt, Vh, out);
}
